// cmMambaWithCNN_51221779972678
// MI455X (gfx1250) — matrix operand order repaired and checked by reading, not compiled, not run on hardware
//
#include <hip/hip_runtime.h>
#include <hip/hip_bf16.h>
#include <stdint.h>

// ---------------------------------------------------------------- types ----
typedef unsigned short u16;
typedef __attribute__((ext_vector_type(8)))  u16     v8u;
typedef __attribute__((ext_vector_type(16))) u16     v16u;
typedef __attribute__((ext_vector_type(16))) __bf16  v16bf;
typedef __attribute__((ext_vector_type(8)))  float   v8f;

// explicit global (AS1) pointer types: keep hot loads as global_load_* (not
// flat_*) even through loop-carried pointer PHIs. FLAT ties up DScnt + the
// LDS path on CDNA5; GLOBAL uses only LOADcnt/STOREcnt.
#define AS1 __attribute__((address_space(1)))
typedef AS1 const u16    gcu16;
typedef AS1 u16          gu16;
typedef AS1 const v8u    gcv8u;
typedef AS1 const v16u   gcv16u;
typedef AS1 const float  gcfloat;
typedef AS1 float        gfloat;
typedef AS1 const int    gcint;

#define LSEQ 4096
#define MTOT 8192
static const float BN_INV = 0.99999500003749969f; // 1/sqrt(1+1e-5)

__device__ __forceinline__ u16 f2bf(float f) {
    union { float f; unsigned u; } c; c.f = f;
    unsigned u = c.u;
    unsigned r = u + 0x7fffu + ((u >> 16) & 1u);   // round-to-nearest-even
    return (u16)(r >> 16);
}

// ------------------------------------------------------------- epilogue ----
struct Epi {
    const float* scale;   // per-col scale (BN gamma); null -> 1
    const float* shift;   // per-col shift (BN beta / bias); null -> 0
    const float* addf;    // residual add, row-major [m*ldd + col]; null -> none
    float*       out_f;   // f32 store (row-major or NCHW)
    u16*         out_h;   // bf16 store (row-major)
    const int*   rowmap;  // out-row permutation (despiral scatter); null -> id
    float scale_mul;
    int ldd, ch_off, relu, nchw;
};

__device__ __forceinline__ void epi_store(const v8f& a, int mrow, int col, int N,
                                          const Epi& e) {
    if (col >= N) return;
    gcfloat* scal = (gcfloat*)e.scale;
    gcfloat* shft = (gcfloat*)e.shift;
    gcfloat* addf = (gcfloat*)e.addf;
    gfloat*  outf = (gfloat*)e.out_f;
    gu16*    outh = (gu16*)e.out_h;
    gcint*   rmap = (gcint*)e.rowmap;
    float sc = scal ? scal[col] * e.scale_mul : 1.0f;
    float sh = shft ? shft[col] : 0.0f;
#pragma unroll
    for (int r = 0; r < 8; r++) {
        int m = mrow + r;
        float v = a[r] * sc + sh;
        if (addf) v += addf[(size_t)m * e.ldd + col];
        if (e.relu) v = fmaxf(v, 0.0f);
        if (e.nchw) {  // final NCHW f32 store: out[(b*ldd + col)*L + p]
            int b = m >> 12, p = m & 4095;
            outf[((size_t)(b * e.ldd + e.ch_off + col)) * LSEQ + p] = v;
        } else {
            size_t row = rmap ? (size_t)rmap[m] : (size_t)m;
            size_t idx = row * (size_t)e.ldd + e.ch_off + col;
            if (outf) outf[idx] = v;
            if (outh) outh[idx] = f2bf(v);
        }
    }
}

// ------------------------------------------------- generic bf16 WMMA GEMM --
// C[M,N] = A[M,K]*B[N,K]^T ; block = 8 waves (2Mx4N), 128x128 tile;
// wave = 64x32 (4x2 WMMA tiles). Branchless: OOB B columns are clamped and
// their (never-stored) accumulators hold garbage.
__global__ __launch_bounds__(256) void k_gemm(
        const u16* __restrict__ A, const u16* __restrict__ Bw,
        int N, int K, int lda, int ldb, const int* __restrict__ a_rowmap, Epi e) {
    const int lane = threadIdx.x & 31, wave = threadIdx.x >> 5;
    const int lo = lane & 15, hi = lane >> 4;
    const int m0 = blockIdx.y * 128 + (wave >> 2) * 64;
    const int n0 = blockIdx.x * 128 + (wave & 3) * 32;
    gcint* rmapA = (gcint*)a_rowmap;

    v8f acc[4][2];
#pragma unroll
    for (int i = 0; i < 4; i++)
#pragma unroll
        for (int j = 0; j < 2; j++)
#pragma unroll
            for (int r = 0; r < 8; r++) acc[i][j][r] = 0.0f;

    gcu16* pa[4];
#pragma unroll
    for (int i = 0; i < 4; i++) {
        int m = m0 + i * 16 + lo;
        pa[i] = (gcu16*)A + (size_t)(rmapA ? rmapA[m] : m) * lda + hi * 16;
    }
    gcu16* pb[2];
#pragma unroll
    for (int j = 0; j < 2; j++) {
        int c = n0 + j * 16 + lo;
        c = (c < N) ? c : (N - 1);          // clamp: keep loads in-bounds
        pb[j] = (gcu16*)Bw + (size_t)c * ldb + hi * 16;
    }

    for (int k0 = 0; k0 < K; k0 += 32) {
        v16u af[4], bfv[2];
#pragma unroll
        for (int i = 0; i < 4; i++) {
            v8u x0 = *(gcv8u*)pa[i];
            v8u x1 = *(gcv8u*)(pa[i] + 8);
#pragma unroll
            for (int t = 0; t < 8; t++) { af[i][t] = x0[t]; af[i][t + 8] = x1[t]; }
            pa[i] += 32;
        }
#pragma unroll
        for (int j = 0; j < 2; j++) {
            bfv[j] = *(gcv16u*)pb[j];
            pb[j] += 32;
        }
#pragma unroll
        for (int i = 0; i < 4; i++)
#pragma unroll
            for (int j = 0; j < 2; j++)
                acc[i][j] = __builtin_amdgcn_wmma_f32_16x16x32_bf16(
                    false, __builtin_bit_cast(v16bf, af[i]),
                    false, __builtin_bit_cast(v16bf, bfv[j]),
                    (short)0, acc[i][j], false, false);
    }
#pragma unroll
    for (int i = 0; i < 4; i++)
#pragma unroll
        for (int j = 0; j < 2; j++)
            epi_store(acc[i][j], m0 + i * 16 + 8 * hi, n0 + j * 16 + lo, N, e);
}

// ----------------------------------------- implicit-GEMM 3x3 conv (NHWC) ---
// In: NHWC bf16 (2,64,64,Cin); Bw: (N, 9*Cin) reordered (o,ky,kx,c).
// Zero-padding handled branchlessly: OOB pixels read from a zeroed page.
__global__ __launch_bounds__(256) void k_conv3(
        const u16* __restrict__ In, int Cin,
        const u16* __restrict__ Bw, int N, int reflectPad,
        const u16* __restrict__ zpage, Epi e) {
    const int lane = threadIdx.x & 31, wave = threadIdx.x >> 5;
    const int lo = lane & 15, hi = lane >> 4;
    const int m0 = blockIdx.y * 128 + (wave >> 2) * 64;
    const int n0 = blockIdx.x * 128 + (wave & 3) * 32;
    const int ldb = 9 * Cin;

    v8f acc[4][2];
#pragma unroll
    for (int i = 0; i < 4; i++)
#pragma unroll
        for (int j = 0; j < 2; j++)
#pragma unroll
            for (int r = 0; r < 8; r++) acc[i][j][r] = 0.0f;

    gcu16* pbase[2];
#pragma unroll
    for (int j = 0; j < 2; j++) {
        int c = n0 + j * 16 + lo;
        c = (c < N) ? c : (N - 1);
        pbase[j] = (gcu16*)Bw + (size_t)c * ldb + hi * 16;
    }
    int mb[4], my[4], mx[4];
#pragma unroll
    for (int i = 0; i < 4; i++) {
        int m = m0 + i * 16 + lo;
        mb[i] = m >> 12; int p = m & 4095; my[i] = p >> 6; mx[i] = p & 63;
    }

    for (int ky = 0; ky < 3; ky++) {
        for (int kx = 0; kx < 3; kx++) {
            gcu16* pa[4];
#pragma unroll
            for (int i = 0; i < 4; i++) {
                int yy = my[i] + ky - 1, xx = mx[i] + kx - 1;
                int ok;
                if (reflectPad) {
                    yy = (yy < 0) ? -yy : ((yy > 63) ? 126 - yy : yy);
                    xx = (xx < 0) ? -xx : ((xx > 63) ? 126 - xx : xx);
                    ok = 1;
                } else {
                    ok = (yy >= 0) & (yy < 64) & (xx >= 0) & (xx < 64);
                }
                const u16* src = ok ? (In + ((size_t)(mb[i] * 4096 + yy * 64 + xx)) * Cin)
                                    : zpage;
                pa[i] = (gcu16*)src + hi * 16;
            }
            gcu16* pbt[2];
#pragma unroll
            for (int j = 0; j < 2; j++) pbt[j] = pbase[j] + (ky * 3 + kx) * Cin;

            for (int c0 = 0; c0 < Cin; c0 += 32) {
                v16u af[4], bfv[2];
#pragma unroll
                for (int i = 0; i < 4; i++) {
                    v8u x0 = *(gcv8u*)pa[i];
                    v8u x1 = *(gcv8u*)(pa[i] + 8);
#pragma unroll
                    for (int t = 0; t < 8; t++) { af[i][t] = x0[t]; af[i][t + 8] = x1[t]; }
                    pa[i] += 32;
                }
#pragma unroll
                for (int j = 0; j < 2; j++) { bfv[j] = *(gcv16u*)pbt[j]; pbt[j] += 32; }
#pragma unroll
                for (int i = 0; i < 4; i++)
#pragma unroll
                    for (int j = 0; j < 2; j++)
                        acc[i][j] = __builtin_amdgcn_wmma_f32_16x16x32_bf16(
                            false, __builtin_bit_cast(v16bf, af[i]),
                            false, __builtin_bit_cast(v16bf, bfv[j]),
                            (short)0, acc[i][j], false, false);
            }
        }
    }
#pragma unroll
    for (int i = 0; i < 4; i++)
#pragma unroll
        for (int j = 0; j < 2; j++)
            epi_store(acc[i][j], m0 + i * 16 + 8 * hi, n0 + j * 16 + lo, N, e);
}

// ------------------------------------------------------- small VALU kernels
__global__ void k_order(const float* __restrict__ S, int* __restrict__ order) {
    int i = blockIdx.x;  // 4096 rows
    for (int j = threadIdx.x; j < LSEQ; j += blockDim.x)
        if (S[(size_t)i * LSEQ + j] > 0.5f) order[i] = j;
}

__global__ void k_rowmaps(const int* __restrict__ order, int* rf, int* rr, int* ar) {
    int m = blockIdx.x * blockDim.x + threadIdx.x;
    if (m >= MTOT) return;
    int b = m >> 12, l = m & 4095;
    rf[m] = b * LSEQ + order[l];
    rr[m] = b * LSEQ + order[4095 - l];
    ar[m] = b * LSEQ + (4095 - l);
}

__global__ void k_zero16(u16* __restrict__ p, int n) {
    int i = blockIdx.x * blockDim.x + threadIdx.x;
    if (i < n) p[i] = 0;
}

__global__ void k_cvt(const float* __restrict__ s, u16* __restrict__ d, int n) {
    int i = blockIdx.x * blockDim.x + threadIdx.x;
    if (i < n) d[i] = f2bf(s[i]);
}

// (O,Cin,3,3) f32 -> (O, 3,3, Cin) bf16
__global__ void k_rw3(const float* __restrict__ s, u16* __restrict__ d, int O, int Cin) {
    int i = blockIdx.x * blockDim.x + threadIdx.x;
    int tot = O * Cin * 9;
    if (i >= tot) return;
    int kk = i % 9; int t = i / 9; int c = t % Cin; int o = t / Cin;
    d[((size_t)o * 9 + kk) * Cin + c] = f2bf(s[i]);
}

// gather (spiral) + LayerNorm over C=256, write bf16 (b,l,c)
__global__ __launch_bounds__(256) void k_ln(
        const float* __restrict__ x, const int* __restrict__ order,
        const float* __restrict__ g, const float* __restrict__ bb,
        u16* __restrict__ xln) {
    __shared__ float red[256];
    int bi = blockIdx.x; int b = bi >> 12, i = bi & 4095; int c = threadIdx.x;
    int j = order[i];
    float v = x[(((size_t)(b * 256 + c)) << 12) + j];
    red[c] = v; __syncthreads();
    for (int s = 128; s > 0; s >>= 1) { if (c < s) red[c] += red[c + s]; __syncthreads(); }
    float mean = red[0] * (1.0f / 256.0f); __syncthreads();
    float dv = v - mean;
    red[c] = dv * dv; __syncthreads();
    for (int s = 128; s > 0; s >>= 1) { if (c < s) red[c] += red[c + s]; __syncthreads(); }
    float var = red[0] * (1.0f / 256.0f);
    xln[((size_t)bi) * 256 + c] = f2bf(dv * rsqrtf(var + 1e-5f) * g[c] + bb[c]);
}

// depthwise causal conv1d (k=4) + bias + silu; reads xc = xz[...,:512]
__global__ void k_conv1d(const float* __restrict__ xz, const float* __restrict__ wc,
                         const float* __restrict__ bc,
                         float* __restrict__ uf, u16* __restrict__ uh) {
    int t = blockIdx.x * blockDim.x + threadIdx.x;
    if (t >= MTOT * 512) return;
    int d = t & 511; int bl = t >> 9; int l = bl & 4095; int b = bl >> 12;
    float acc = bc[d];
#pragma unroll
    for (int k = 0; k < 4; k++) {
        int ls = l - 3 + k;
        if (ls >= 0) acc += xz[(((size_t)(b * LSEQ + ls)) << 10) + d] * wc[d * 4 + k];
    }
    float s = acc / (1.0f + __expf(-acc));
    uf[t] = s; uh[t] = f2bf(s);
}

// dt = softplus(dt_part @ W_dt^T + b_dt)   (K=16: VALU)
__global__ void k_dt(const float* __restrict__ xdbl, const float* __restrict__ wdt,
                     const float* __restrict__ bdt, float* __restrict__ dtb) {
    int t = blockIdx.x * blockDim.x + threadIdx.x;
    if (t >= MTOT * 512) return;
    int d = t & 511; int bl = t >> 9;
    const float* row = xdbl + (size_t)bl * 48;
    float acc = bdt[d];
#pragma unroll
    for (int r = 0; r < 16; r++) acc += row[r] * wdt[d * 16 + r];
    dtb[t] = (acc > 20.0f) ? acc : log1pf(__expf(acc));
}

// selective scan: thread = (d,n); 16-lane shuffle reduce over n; fuse gate.
// All streams walked with incrementing global-AS pointers.
__global__ __launch_bounds__(256) void k_scan(
        const float* __restrict__ dtb, const float* __restrict__ uf,
        const float* __restrict__ xdbl, const float* __restrict__ xz,
        const float* __restrict__ Alog, const float* __restrict__ Dp,
        u16* __restrict__ yg) {
    int b = blockIdx.x >> 5; int dblk = blockIdx.x & 31;
    int d = dblk * 16 + (threadIdx.x >> 4); int n = threadIdx.x & 15;
    float A = -__expf(Alog[d * 16 + n]);
    float Dd = Dp[d];
    float h = 0.0f;
    const size_t base = (size_t)b * LSEQ;
    gcfloat* pdt = (gcfloat*)dtb  + (base << 9) + d;
    gcfloat* pu  = (gcfloat*)uf   + (base << 9) + d;
    gcfloat* pB  = (gcfloat*)xdbl + base * 48 + 16 + n;
    gcfloat* pz  = (gcfloat*)xz   + (base << 10) + 512 + d;
    gu16*    py  = (gu16*)yg      + (base << 9) + d;
    for (int t = 0; t < LSEQ; t++) {
        float dt = *pdt; pdt += 512;
        float u  = *pu;  pu  += 512;
        float Bv = pB[0];
        float Cv = pB[16]; pB += 48;
        h = h * __expf(dt * A) + (dt * u) * Bv;
        float p = h * Cv;
        p += __shfl_xor(p, 1);
        p += __shfl_xor(p, 2);
        p += __shfl_xor(p, 4);
        p += __shfl_xor(p, 8);
        if (n == 0) {
            float z = *pz;
            float v = (p + u * Dd) * (z / (1.0f + __expf(-z)));
            *py = f2bf(v);
        }
        pz += 1024; py += 512;
    }
}

// x (NCHW f32) -> xcat NHWC bf16, channels 0..255
__global__ void k_xcat(const float* __restrict__ x, u16* __restrict__ xcat) {
    int t = blockIdx.x * blockDim.x + threadIdx.x;
    if (t >= MTOT * 256) return;
    int c = t & 255; int bp = t >> 8; int p = bp & 4095; int b = bp >> 12;
    xcat[((size_t)bp) * 512 + c] = f2bf(x[(((size_t)(b * 256 + c)) << 12) + p]);
}

// ----------------------------------------------------------------- launch --
static Epi mkEpi() {
    Epi e; e.scale = nullptr; e.shift = nullptr; e.addf = nullptr;
    e.out_f = nullptr; e.out_h = nullptr; e.rowmap = nullptr;
    e.scale_mul = 1.0f; e.ldd = 0; e.ch_off = 0; e.relu = 0; e.nchw = 0;
    return e;
}

extern "C" void kernel_launch(void* const* d_in, const int* in_sizes, int n_in,
                              void* d_out, int out_size, void* d_ws, size_t ws_size,
                              hipStream_t stream) {
    (void)in_sizes; (void)n_in; (void)out_size; (void)ws_size;
    const float* x    = (const float*)d_in[0];
    const float* S    = (const float*)d_in[1];
    const float* ln_w = (const float*)d_in[4];
    const float* ln_b = (const float*)d_in[5];
    const float* Win[2]   = {(const float*)d_in[6],  (const float*)d_in[15]};
    const float* Wconv[2] = {(const float*)d_in[7],  (const float*)d_in[16]};
    const float* bconv[2] = {(const float*)d_in[8],  (const float*)d_in[17]};
    const float* Wx[2]    = {(const float*)d_in[9],  (const float*)d_in[18]};
    const float* Wdt[2]   = {(const float*)d_in[10], (const float*)d_in[19]};
    const float* bdt[2]   = {(const float*)d_in[11], (const float*)d_in[20]};
    const float* Alog[2]  = {(const float*)d_in[12], (const float*)d_in[21]};
    const float* Dvec[2]  = {(const float*)d_in[13], (const float*)d_in[22]};
    const float* Wout[2]  = {(const float*)d_in[14], (const float*)d_in[23]};
    const float* Wfuse = (const float*)d_in[24]; const float* bfuse = (const float*)d_in[25];
    const float* Wskip = (const float*)d_in[26];
    const float* gskip = (const float*)d_in[27]; const float* btskip = (const float*)d_in[28];
    const float* Wcc1 = (const float*)d_in[29];
    const float* gcc1 = (const float*)d_in[30]; const float* btcc1 = (const float*)d_in[31];
    const float* Wcc2 = (const float*)d_in[32];
    const float* gcc2 = (const float*)d_in[33]; const float* btcc2 = (const float*)d_in[34];
    const float* Wr1 = (const float*)d_in[35];
    const float* gr1 = (const float*)d_in[36]; const float* btr1 = (const float*)d_in[37];
    const float* Wr2 = (const float*)d_in[38];
    const float* gr2 = (const float*)d_in[39]; const float* btr2 = (const float*)d_in[40];

    // workspace carve-out (256B aligned)
    char* W = (char*)d_ws; size_t off = 0;
    auto ALLOC = [&](size_t bytes) -> void* {
        off = (off + 255) & ~(size_t)255; void* p = W + off; off += bytes; return p;
    };
    int* order = (int*)ALLOC(LSEQ * 4);
    int* rmf = (int*)ALLOC(MTOT * 4);
    int* rmr = (int*)ALLOC(MTOT * 4);
    int* arv = (int*)ALLOC(MTOT * 4);
    u16* zpage = (u16*)ALLOC(1024 * 2);          // zero page for conv padding
    u16* wb_in[2]  = {(u16*)ALLOC(262144 * 2), (u16*)ALLOC(262144 * 2)};
    u16* wb_x[2]   = {(u16*)ALLOC(24576 * 2),  (u16*)ALLOC(24576 * 2)};
    u16* wb_out[2] = {(u16*)ALLOC(131072 * 2), (u16*)ALLOC(131072 * 2)};
    u16* wb_fuse = (u16*)ALLOC(131072 * 2);
    u16* wb_skip = (u16*)ALLOC(131072 * 2);
    u16* wb_cc1 = (u16*)ALLOC(1179648 * 2);
    u16* wb_cc2 = (u16*)ALLOC(589824 * 2);
    u16* wb_r1  = (u16*)ALLOC(589824 * 2);
    u16* wb_r2  = (u16*)ALLOC(589824 * 2);
    u16* xln    = (u16*)ALLOC((size_t)MTOT * 256 * 2);
    u16* outcat = (u16*)ALLOC((size_t)MTOT * 512 * 2);
    u16* xcat   = (u16*)ALLOC((size_t)MTOT * 512 * 2);
    u16* blk1   = (u16*)ALLOC((size_t)MTOT * 256 * 2);
    u16* yb     = (u16*)ALLOC((size_t)MTOT * 256 * 2);
    u16* r1b    = (u16*)ALLOC((size_t)MTOT * 256 * 2);
    float* skipb = (float*)ALLOC((size_t)MTOT * 256 * 4);
    float* yf    = (float*)ALLOC((size_t)MTOT * 256 * 4);
    float* xz    = (float*)ALLOC((size_t)MTOT * 1024 * 4);
    float* uf    = (float*)ALLOC((size_t)MTOT * 512 * 4);
    u16*   ub    = (u16*)ALLOC((size_t)MTOT * 512 * 2);
    float* xdbl  = (float*)ALLOC((size_t)MTOT * 48 * 4);
    float* dtb   = (float*)ALLOC((size_t)MTOT * 512 * 4);
    u16*   yg    = (u16*)ALLOC((size_t)MTOT * 512 * 2);

    // --- prep: permutation, rowmaps, zero page, weight conversions --------
    k_order<<<LSEQ, 256, 0, stream>>>(S, order);
    k_rowmaps<<<(MTOT + 255) / 256, 256, 0, stream>>>(order, rmf, rmr, arv);
    k_zero16<<<4, 256, 0, stream>>>(zpage, 1024);
    for (int d = 0; d < 2; d++) {
        k_cvt<<<(262144 + 255) / 256, 256, 0, stream>>>(Win[d], wb_in[d], 262144);
        k_cvt<<<(24576 + 255) / 256, 256, 0, stream>>>(Wx[d], wb_x[d], 24576);
        k_cvt<<<(131072 + 255) / 256, 256, 0, stream>>>(Wout[d], wb_out[d], 131072);
    }
    k_cvt<<<(131072 + 255) / 256, 256, 0, stream>>>(Wfuse, wb_fuse, 131072);
    k_cvt<<<(131072 + 255) / 256, 256, 0, stream>>>(Wskip, wb_skip, 131072);
    k_rw3<<<(1179648 + 255) / 256, 256, 0, stream>>>(Wcc1, wb_cc1, 256, 512);
    k_rw3<<<(589824 + 255) / 256, 256, 0, stream>>>(Wcc2, wb_cc2, 256, 256);
    k_rw3<<<(589824 + 255) / 256, 256, 0, stream>>>(Wr1, wb_r1, 256, 256);
    k_rw3<<<(589824 + 255) / 256, 256, 0, stream>>>(Wr2, wb_r2, 256, 256);

    // --- gather + layernorm (shared by both directions) ------------------
    k_ln<<<MTOT, 256, 0, stream>>>(x, order, ln_w, ln_b, xln);

    // --- two mamba branches ----------------------------------------------
    const dim3 blk(256);
    for (int d = 0; d < 2; d++) {
        Epi e1 = mkEpi(); e1.out_f = xz; e1.ldd = 1024;
        k_gemm<<<dim3(8, MTOT / 128), blk, 0, stream>>>(
            xln, wb_in[d], 1024, 256, 256, 256, d ? arv : nullptr, e1);

        k_conv1d<<<(MTOT * 512 + 255) / 256, 256, 0, stream>>>(xz, Wconv[d], bconv[d], uf, ub);

        Epi e2 = mkEpi(); e2.out_f = xdbl; e2.ldd = 48;
        k_gemm<<<dim3(1, MTOT / 128), blk, 0, stream>>>(ub, wb_x[d], 48, 512, 512, 512, nullptr, e2);

        k_dt<<<(MTOT * 512 + 255) / 256, 256, 0, stream>>>(xdbl, Wdt[d], bdt[d], dtb);

        k_scan<<<64, 256, 0, stream>>>(dtb, uf, xdbl, xz, Alog[d], Dvec[d], yg);

        Epi e3 = mkEpi(); e3.out_h = outcat; e3.ldd = 512; e3.ch_off = d * 256;
        e3.rowmap = d ? rmr : rmf;  // despiral scatter
        k_gemm<<<dim3(2, MTOT / 128), blk, 0, stream>>>(
            yg, wb_out[d], 256, 512, 512, 512, nullptr, e3);
    }

    // --- CNN head ---------------------------------------------------------
    k_xcat<<<(MTOT * 256 + 255) / 256, 256, 0, stream>>>(x, xcat);

    Epi ef = mkEpi(); ef.shift = bfuse; ef.out_h = xcat; ef.ldd = 512; ef.ch_off = 256;
    k_gemm<<<dim3(2, MTOT / 128), blk, 0, stream>>>(outcat, wb_fuse, 256, 512, 512, 512, nullptr, ef);

    Epi es = mkEpi(); es.scale = gskip; es.scale_mul = BN_INV; es.shift = btskip;
    es.out_f = skipb; es.ldd = 256;
    k_gemm<<<dim3(2, MTOT / 128), blk, 0, stream>>>(xcat, wb_skip, 256, 512, 512, 512, nullptr, es);

    Epi ec1 = mkEpi(); ec1.scale = gcc1; ec1.scale_mul = BN_INV; ec1.shift = btcc1;
    ec1.relu = 1; ec1.out_h = blk1; ec1.ldd = 256;
    k_conv3<<<dim3(2, MTOT / 128), blk, 0, stream>>>(xcat, 512, wb_cc1, 256, 0, zpage, ec1);

    Epi ec2 = mkEpi(); ec2.scale = gcc2; ec2.scale_mul = BN_INV; ec2.shift = btcc2;
    ec2.addf = skipb; ec2.relu = 1; ec2.out_f = yf; ec2.out_h = yb; ec2.ldd = 256;
    k_conv3<<<dim3(2, MTOT / 128), blk, 0, stream>>>(blk1, 256, wb_cc2, 256, 0, zpage, ec2);

    Epi er1 = mkEpi(); er1.scale = gr1; er1.scale_mul = BN_INV; er1.shift = btr1;
    er1.relu = 1; er1.out_h = r1b; er1.ldd = 256;
    k_conv3<<<dim3(2, MTOT / 128), blk, 0, stream>>>(yb, 256, wb_r1, 256, 1, zpage, er1);

    Epi er2 = mkEpi(); er2.scale = gr2; er2.scale_mul = BN_INV; er2.shift = btr2;
    er2.addf = yf; er2.out_f = (float*)d_out; er2.ldd = 256; er2.nchw = 1;
    k_conv3<<<dim3(2, MTOT / 128), blk, 0, stream>>>(r1b, 256, wb_r2, 256, 1, zpage, er2);
}
